// MultiHeadAttention_88201448391332
// MI455X (gfx1250) — matrix operand order repaired and checked by reading, not compiled, not run on hardware
//
#include <hip/hip_runtime.h>
#include <hip/hip_bf16.h>

typedef __attribute__((ext_vector_type(16))) __bf16 v16bf;
typedef __attribute__((ext_vector_type(8)))  float  v8f;

#define DM     1024
#define S_LEN  2048
#define NHEADS 16
#define DK     64
#define BATCH  4

// float -> bf16, round-to-nearest-even
__device__ __forceinline__ __bf16 f2bf(float x) {
  union { float f; unsigned u; } a; a.f = x;
  unsigned r = a.u + 0x7FFFu + ((a.u >> 16) & 1u);
  union { unsigned short s; __bf16 b; } c; c.s = (unsigned short)(r >> 16);
  return c.b;
}

// pack two f32 into one dword of two RNE bf16 (lo in [15:0], hi in [31:16])
__device__ __forceinline__ unsigned pkbf2(float lo, float hi) {
  union { float f; unsigned u; } a, b; a.f = lo; b.f = hi;
  const unsigned rl = (a.u + 0x7FFFu + ((a.u >> 16) & 1u)) >> 16;
  const unsigned rh = (b.u + 0x7FFFu + ((b.u >> 16) & 1u)) & 0xFFFF0000u;
  return rh | rl;
}

union Frag32B { v16bf v; uint4 u[2]; };

__device__ __forceinline__ v8f zero8() {
  v8f z = {0.f,0.f,0.f,0.f,0.f,0.f,0.f,0.f};
  return z;
}

__device__ __forceinline__ v8f wmma_bf16(v16bf a, v16bf b, v8f c) {
  // D = A(16x32 bf16) x B(32x16 bf16) + C(16x16 f32)
  return __builtin_amdgcn_wmma_f32_16x16x32_bf16(false, a, false, b, (short)0, c,
                                                 false, false);
}

// ---- fragment builders (bf16 sources only; all 16B-aligned b128 loads) ------
// A fragment (16x32): lane L -> row M=L%16; half h=L/16:
//   elements 0..7  -> K = h*8 + (0..7)        elements 8..15 -> K = 16 + h*8 + (0..7)
__device__ __forceinline__ v16bf fragA_bf16(const __bf16* p, int hlf) {
  Frag32B f;
  f.u[0] = *(const uint4*)(p + hlf * 8);
  f.u[1] = *(const uint4*)(p + 16 + hlf * 8);
  return f.v;
}
// B fragment (32x16): lane L -> col N=L%16; half h=L/16:
//   elements 0..7  -> K = h*8 + (0..7)        elements 8..15 -> K = 16 + h*8 + (0..7)
__device__ __forceinline__ v16bf fragB_bf16(const __bf16* p, int hlf) {
  Frag32B f;
  f.u[0] = *(const uint4*)(p + hlf * 8);
  f.u[1] = *(const uint4*)(p + 16 + hlf * 8);
  return f.v;
}

// ---- bulk f32 -> bf16 conversion (one pass; 8 elements per thread) ----------
__global__ __launch_bounds__(256) void cvt_bf16(const float* __restrict__ in,
                                                unsigned* __restrict__ out,
                                                int n8) {
  const int i = (int)(blockIdx.x * blockDim.x + threadIdx.x);
  if (i >= n8) return;
  const float4 f0 = ((const float4*)in)[i * 2 + 0];
  const float4 f1 = ((const float4*)in)[i * 2 + 1];
  uint4 o;
  o.x = pkbf2(f0.x, f0.y);
  o.y = pkbf2(f0.z, f0.w);
  o.z = pkbf2(f1.x, f1.y);
  o.w = pkbf2(f1.z, f1.w);
  ((uint4*)out)[i] = o;
}

// ---- GEMM: C[M,N] = A[M,K] * W[N,K]^T  (per-wave 64x64 tile, 16 WMMA / K-step)
// Both A and W are bf16; inner loop is pure b128 loads + WMMA.
// OMODE: 0 = f32 out, 1 = bf16 out, 2 = bf16 out * 0.125 (q), 3 = bf16 vT store
template<int OMODE>
__global__ __launch_bounds__(256) void gemm_xwt(const __bf16* __restrict__ Ap,
                                                const __bf16* __restrict__ W,
                                                void* __restrict__ Outp,
                                                int M, int N, int K) {
  const int lane = threadIdx.x & 31;
  const int l16  = lane & 15;
  const int hlf  = lane >> 4;
  const int wv   = (int)((blockIdx.x * blockDim.x + threadIdx.x) >> 5);
  const int ntiles = N >> 6;
  const int mt = wv / ntiles;
  const int nt = wv - mt * ntiles;
  if (mt * 64 >= M) return;

  v8f acc[4][4];
  #pragma unroll
  for (int i = 0; i < 4; ++i)
    #pragma unroll
    for (int j = 0; j < 4; ++j) acc[i][j] = zero8();

  const __bf16* arow = Ap + (size_t)(mt * 64 + l16) * K;
  const __bf16* wrow = W  + (size_t)(nt * 64 + l16) * K;

  for (int k0 = 0; k0 < K; k0 += 32) {
    v16bf aF[4], bF[4];
    #pragma unroll
    for (int mb = 0; mb < 4; ++mb)
      aF[mb] = fragA_bf16(arow + (size_t)(mb * 16) * K + k0, hlf);
    #pragma unroll
    for (int nb = 0; nb < 4; ++nb)
      bF[nb] = fragB_bf16(wrow + (size_t)(nb * 16) * K + k0, hlf);
    #pragma unroll
    for (int mb = 0; mb < 4; ++mb)
      #pragma unroll
      for (int nb = 0; nb < 4; ++nb)
        acc[mb][nb] = wmma_bf16(aF[mb], bF[nb], acc[mb][nb]);
  }

  // C/D layout: VGPR r, lane L -> row = r + 8*(L/16), col = L%16
  #pragma unroll
  for (int mb = 0; mb < 4; ++mb) {
    #pragma unroll
    for (int nb = 0; nb < 4; ++nb) {
      #pragma unroll
      for (int r = 0; r < 8; ++r) {
        const int row = mt * 64 + mb * 16 + r + 8 * hlf;
        const int col = nt * 64 + nb * 16 + l16;
        const float v = acc[mb][nb][r];
        if (OMODE == 0) {
          ((float*)Outp)[(size_t)row * N + col] = v;
        } else if (OMODE == 1) {
          ((__bf16*)Outp)[(size_t)row * N + col] = f2bf(v);
        } else if (OMODE == 2) {
          ((__bf16*)Outp)[(size_t)row * N + col] = f2bf(v * 0.125f);
        } else {
          const int bb = row >> 11, s = row & 2047;  // row = b*2048 + s
          const int hh = col >> 6,  d = col & 63;    // col = h*64 + d
          ((__bf16*)Outp)[((size_t)((bb * NHEADS + hh) * DK + d) << 11) + s] = f2bf(v);
        }
      }
    }
  }
}

// ---- Flash attention, one wave per (b, h, 16-query tile) --------------------
// Scores computed transposed: T = K_tile(32x64) x q^T(64x16) so each query is a
// lane-column; softmax is per-lane + one shfl_xor(16); P re-enters as A-frag
// with no cross-lane traffic.
__global__ __launch_bounds__(256) void flash_attn(const __bf16* __restrict__ qh,
                                                  const __bf16* __restrict__ kh,
                                                  const __bf16* __restrict__ vT,
                                                  __bf16* __restrict__ ho) {
  const int lane = threadIdx.x & 31;
  const int l16  = lane & 15;
  const int hlf  = lane >> 4;
  const int wv = (int)((blockIdx.x * blockDim.x + threadIdx.x) >> 5);
  const int qt = wv & 127;          // S/16 = 128 query tiles
  const int h  = (wv >> 7) & 15;
  const int b  = wv >> 11;

  // q^T B-fragments (d = K dim), preloaded; q already scaled by 0.125 at store.
  const __bf16* qrowp = qh + (size_t)(b * S_LEN + qt * 16 + l16) * DM + h * DK;
  v16bf bq0 = fragB_bf16(qrowp +  0, hlf);
  v16bf bq1 = fragB_bf16(qrowp + 32, hlf);

  const __bf16* kcol = kh + (size_t)b * S_LEN * DM + h * DK;        // + key*DM
  const __bf16* vhd  = vT + (size_t)(b * NHEADS + h) * DK * S_LEN;  // [64][2048]

  v8f o[4];
  #pragma unroll
  for (int db = 0; db < 4; ++db) o[db] = zero8();
  float m_run = -3.0e30f, l_run = 0.f;

  for (int kt = 0; kt < S_LEN / 32; ++kt) {
    const int keyb = kt * 32;
    const __bf16* kp0 = kcol + (size_t)(keyb + l16) * DM;
    const __bf16* kp1 = kcol + (size_t)(keyb + 16 + l16) * DM;

    v8f t0 = zero8(), t1 = zero8();
    t0 = wmma_bf16(fragA_bf16(kp0 +  0, hlf), bq0, t0);
    t0 = wmma_bf16(fragA_bf16(kp0 + 32, hlf), bq1, t0);
    t1 = wmma_bf16(fragA_bf16(kp1 +  0, hlf), bq0, t1);
    t1 = wmma_bf16(fragA_bf16(kp1 + 32, hlf), bq1, t1);

    // online softmax over this lane's 32 key-scores for its query column
    float mx = -3.0e30f;
    #pragma unroll
    for (int r = 0; r < 8; ++r) mx = fmaxf(mx, fmaxf(t0[r], t1[r]));
    mx = fmaxf(mx, __shfl_xor(mx, 16, 32));
    const float mnew  = fmaxf(m_run, mx);
    const float alpha = __expf(m_run - mnew);

    Frag32B pf;
    float sum = 0.f;
    #pragma unroll
    for (int r = 0; r < 8; ++r) {
      const float e0 = __expf(t0[r] - mnew);
      const float e1 = __expf(t1[r] - mnew);
      sum += e0 + e1;
      // A-frag(16x32): elem r -> K = 8h + r (T0 rows), elem 8+r -> K = 16+8h+r (T1)
      pf.v[r]     = f2bf(e0);
      pf.v[8 + r] = f2bf(e1);
    }
    sum += __shfl_xor(sum, 16, 32);
    l_run = l_run * alpha + sum;
    m_run = mnew;

    // rescale O accumulators: row of O = query = r + 8h; alpha lives in lane q
    #pragma unroll
    for (int r = 0; r < 8; ++r) {
      const float ar = __shfl(alpha, hlf * 8 + r, 32);
      #pragma unroll
      for (int db = 0; db < 4; ++db) o[db][r] *= ar;
    }
    // O += P(16x32) x V(32x16 per d-block); vT rows are contiguous in key dim
    #pragma unroll
    for (int db = 0; db < 4; ++db) {
      const __bf16* vp = vhd + (size_t)(db * 16 + l16) * S_LEN + keyb;
      o[db] = wmma_bf16(pf.v, fragB_bf16(vp, hlf), o[db]);
    }
  }

  const float linv = 1.0f / l_run;
  __bf16* orow = ho + (size_t)(b * S_LEN + qt * 16) * DM + h * DK;
  #pragma unroll
  for (int r = 0; r < 8; ++r) {
    const float lr = __shfl(linv, hlf * 8 + r, 32);
    const int row = r + 8 * hlf;
    #pragma unroll
    for (int db = 0; db < 4; ++db)
      orow[(size_t)row * DM + db * 16 + l16] = f2bf(o[db][r] * lr);
  }
}

extern "C" void kernel_launch(void* const* d_in, const int* in_sizes, int n_in,
                              void* d_out, int out_size, void* d_ws, size_t ws_size,
                              hipStream_t stream) {
  (void)in_sizes; (void)n_in; (void)out_size; (void)ws_size;
  const float* Q  = (const float*)d_in[0];
  const float* Kx = (const float*)d_in[1];
  const float* V  = (const float*)d_in[2];
  const float* WQ = (const float*)d_in[3];
  const float* WK = (const float*)d_in[4];
  const float* WV = (const float*)d_in[5];
  const float* WO = (const float*)d_in[6];

  const size_t MN = (size_t)BATCH * S_LEN * DM;  // 8,388,608 elements
  const size_t WN = (size_t)DM * DM;             // 1,048,576 elements
  __bf16* p = (__bf16*)d_ws;
  __bf16* Qb  = p; p += MN;   // converted inputs
  __bf16* Kb  = p; p += MN;
  __bf16* Vb  = p; p += MN;
  __bf16* WQb = p; p += WN;   // converted weights
  __bf16* WKb = p; p += WN;
  __bf16* WVb = p; p += WN;
  __bf16* WOb = p; p += WN;
  __bf16* qhp = p; p += MN;   // [B*S, DM]  q (pre-scaled by 1/sqrt(dk))
  __bf16* khp = p; p += MN;   // [B*S, DM]  k
  __bf16* vTp = p; p += MN;   // [B, H, DK, S]  v transposed per head
  __bf16* hop = p;            // [B*S, DM]  merged attention heads

  dim3 blk(256);
  // one-pass conversions (8 elements / thread)
  dim3 cg_in((unsigned)(MN / 8 / 256));   // 4096 blocks
  dim3 cg_w((unsigned)(WN / 8 / 256));    // 512 blocks
  cvt_bf16<<<cg_in, blk, 0, stream>>>(Q,  (unsigned*)Qb,  (int)(MN / 8));
  cvt_bf16<<<cg_in, blk, 0, stream>>>(Kx, (unsigned*)Kb,  (int)(MN / 8));
  cvt_bf16<<<cg_in, blk, 0, stream>>>(V,  (unsigned*)Vb,  (int)(MN / 8));
  cvt_bf16<<<cg_w,  blk, 0, stream>>>(WQ, (unsigned*)WQb, (int)(WN / 8));
  cvt_bf16<<<cg_w,  blk, 0, stream>>>(WK, (unsigned*)WKb, (int)(WN / 8));
  cvt_bf16<<<cg_w,  blk, 0, stream>>>(WV, (unsigned*)WVb, (int)(WN / 8));
  cvt_bf16<<<cg_w,  blk, 0, stream>>>(WO, (unsigned*)WOb, (int)(WN / 8));

  const int M = BATCH * S_LEN;     // 8192
  // (M/64)*(DM/64) = 2048 waves -> 256 blocks of 8 waves
  dim3 gemm_grid((unsigned)((M / 64) * (DM / 64) * 32 / 256));
  gemm_xwt<2><<<gemm_grid, blk, 0, stream>>>(Qb, WQb, qhp, M, DM, DM);
  gemm_xwt<1><<<gemm_grid, blk, 0, stream>>>(Kb, WKb, khp, M, DM, DM);
  gemm_xwt<3><<<gemm_grid, blk, 0, stream>>>(Vb, WVb, vTp, M, DM, DM);
  // B*H*(S/16) = 8192 waves -> 1024 blocks
  dim3 attn_grid((unsigned)(BATCH * NHEADS * (S_LEN / 16) * 32 / 256));
  flash_attn<<<attn_grid, blk, 0, stream>>>(qhp, khp, vTp, hop);
  gemm_xwt<0><<<gemm_grid, blk, 0, stream>>>(hop, WOb, (float*)d_out, M, DM, DM);
}
